// SCNN_76381698392871
// MI455X (gfx1250) — compile-verified, matrix operand order checked by reading, not run on hardware
//
#include <hip/hip_runtime.h>

// Problem constants (from reference)
#define B_   16
#define C_   128
#define H_   72
#define W_   200
#define KS_  9
#define PAD_ 4
#define HW_  (H_ * W_)

// Tiling
#define TILE_N 32
#define HALO   40   // TILE_N + 2*PAD
#define LROW   136  // C_ + 8 ushort padding -> 272B row stride, bank-conflict free

typedef __attribute__((ext_vector_type(16))) __bf16       v16bf;
typedef __attribute__((ext_vector_type(8)))  float        v8f;
typedef __attribute__((ext_vector_type(4)))  unsigned int v4u;

union Frag32B {
  v16bf bf;
  v4u   q[2];
};

static __device__ __forceinline__ unsigned short f2bf(float f) {
  unsigned int u = __builtin_bit_cast(unsigned int, f);
  u += 0x7fffu + ((u >> 16) & 1u); // round-to-nearest-even
  return (unsigned short)(u >> 16);
}

// Pack fp32 weights [co][ci][k] -> bf16 [k][co][ci] so A-fragments are two
// contiguous 16B loads per lane.
__global__ __launch_bounds__(256) void scnn_pack_w(const float* __restrict__ w,
                                                   unsigned short* __restrict__ wp) {
  int i = blockIdx.x * 256 + threadIdx.x;
  if (i >= C_ * C_ * KS_) return;
  int ci = i % C_;
  int co = (i / C_) % C_;
  int k  = i / (C_ * C_);
  wp[i] = f2bf(w[(co * C_ + ci) * KS_ + k]);
}

__global__ __launch_bounds__(256) void scnn_copy(const float* __restrict__ in,
                                                 float* __restrict__ out, int n4) {
  typedef __attribute__((ext_vector_type(4))) float v4f;
  int i = blockIdx.x * 256 + threadIdx.x;
  if (i < n4) ((v4f*)out)[i] = ((const v4f*)in)[i];
}

// One scan step: out[:, :, pos] += relu( conv1d_k9( out[:, :, prevpos] ) )
// Generic over scan axis via strides:
//   vertical   sweep: nLen=W, nStride=1, sStride=W  (conv along W, scan over H)
//   horizontal sweep: nLen=H, nStride=W, sStride=1  (conv along H, scan over W)
__global__ __launch_bounds__(256) void scnn_step(float* __restrict__ out,
                                                 const unsigned short* __restrict__ wp,
                                                 int pos, int prevpos,
                                                 int nLen, int nStride, int sStride) {
  __shared__ unsigned short lds[HALO * LROW]; // [n-halo][ci] bf16, padded rows

  const int    b        = blockIdx.y;
  const int    n0       = blockIdx.x * TILE_N;
  const int    tid      = threadIdx.x;
  const size_t bBase    = (size_t)b * C_ * HW_;
  const size_t prevBase = bBase + (size_t)prevpos * sStride;

  // Stage previous line (with +-PAD halo, zero-padded) to LDS as bf16,
  // transposed to [n][ci] so B-fragments are contiguous ds_load_b128 pairs.
  for (int idx = tid; idx < HALO * C_; idx += 256) {
    int   ci = idx / HALO;
    int   j  = idx - ci * HALO;
    int   pn = n0 - PAD_ + j;
    float v  = 0.f;
    if (pn >= 0 && pn < nLen)
      v = out[prevBase + (size_t)ci * HW_ + (size_t)pn * nStride];
    lds[j * LROW + ci] = f2bf(v);
  }
  __syncthreads();

  const int wave = tid >> 5;  // 8 waves -> 8 M-tiles of 16 (covers C_out=128)
  const int lane = tid & 31;
  const int lrow = lane & 15;
  const int hi   = lane >> 4;
  const int m0   = wave * 16;

  v8f acc0 = {0.f, 0.f, 0.f, 0.f, 0.f, 0.f, 0.f, 0.f};
  v8f acc1 = acc0;

#pragma unroll
  for (int k = 0; k < KS_; ++k) {
    const unsigned short* wk = wp + ((size_t)k * C_ + (m0 + lrow)) * C_;
    if (k + 1 < KS_) __builtin_prefetch(wk + C_ * C_, 0, 1); // global_prefetch_b8
    const int j0 = (lrow + k) * LROW;        // N-subtile 0, shifted by tap
    const int j1 = (16 + lrow + k) * LROW;   // N-subtile 1
#pragma unroll
    for (int kc = 0; kc < 4; ++kc) {         // K = 128 = 4 x 32
      const int cb = kc * 32 + hi * 8;       // 16-bit A/B lane layout (ISA 7.12.2)
      Frag32B a, b0, b1;
      a.q[0]  = *(const v4u*)(wk + cb);
      a.q[1]  = *(const v4u*)(wk + cb + 16);
      b0.q[0] = *(const v4u*)(lds + j0 + cb);
      b0.q[1] = *(const v4u*)(lds + j0 + cb + 16);
      b1.q[0] = *(const v4u*)(lds + j1 + cb);
      b1.q[1] = *(const v4u*)(lds + j1 + cb + 16);
      acc0 = __builtin_amdgcn_wmma_f32_16x16x32_bf16(false, a.bf, false, b0.bf,
                                                     (short)0, acc0, false, false);
      acc1 = __builtin_amdgcn_wmma_f32_16x16x32_bf16(false, a.bf, false, b1.bf,
                                                     (short)0, acc1, false, false);
    }
  }

  // f32 16x16 C/D layout: lane -> N, VGPR r -> M = 8*(lane>>4) + r
  const size_t curBase = bBase + (size_t)pos * sStride + (size_t)(m0 + hi * 8) * HW_;
  int n_a = n0 + lrow;
  if (n_a < nLen) {
    size_t addr = curBase + (size_t)n_a * nStride;
#pragma unroll
    for (int r = 0; r < 8; ++r) {
      float v = acc0[r];
      out[addr + (size_t)r * HW_] += (v > 0.f ? v : 0.f);
    }
  }
  int n_b = n0 + 16 + lrow;
  if (n_b < nLen) {
    size_t addr = curBase + (size_t)n_b * nStride;
#pragma unroll
    for (int r = 0; r < 8; ++r) {
      float v = acc1[r];
      out[addr + (size_t)r * HW_] += (v > 0.f ? v : 0.f);
    }
  }
}

extern "C" void kernel_launch(void* const* d_in, const int* in_sizes, int n_in,
                              void* d_out, int out_size, void* d_ws, size_t ws_size,
                              hipStream_t stream) {
  (void)in_sizes; (void)n_in; (void)out_size; (void)ws_size;
  const float* x    = (const float*)d_in[0];
  const float* w_ud = (const float*)d_in[1];
  const float* w_du = (const float*)d_in[2];
  const float* w_lr = (const float*)d_in[3];
  const float* w_rl = (const float*)d_in[4];
  float*          out = (float*)d_out;
  unsigned short* wp  = (unsigned short*)d_ws;

  const size_t wElems  = (size_t)C_ * C_ * KS_; // 147456 bf16 per tensor
  const int    pblocks = (int)((wElems + 255) / 256);
  scnn_pack_w<<<pblocks, 256, 0, stream>>>(w_ud, wp + 0 * wElems);
  scnn_pack_w<<<pblocks, 256, 0, stream>>>(w_du, wp + 1 * wElems);
  scnn_pack_w<<<pblocks, 256, 0, stream>>>(w_lr, wp + 2 * wElems);
  scnn_pack_w<<<pblocks, 256, 0, stream>>>(w_rl, wp + 3 * wElems);

  const int nf4 = B_ * C_ * H_ * W_ / 4;
  scnn_copy<<<(nf4 + 255) / 256, 256, 0, stream>>>(x, out, nf4);

  dim3 gridV((W_ + TILE_N - 1) / TILE_N, B_); // 7 x 16 blocks per step
  dim3 gridH((H_ + TILE_N - 1) / TILE_N, B_); // 3 x 16 blocks per step

  // Pass 1: top-down (scan H, conv along W)
  for (int i = 1; i < H_; ++i)
    scnn_step<<<gridV, 256, 0, stream>>>(out, wp + 0 * wElems, i, i - 1, W_, 1, W_);
  // Pass 2: bottom-up
  for (int i = H_ - 2; i >= 0; --i)
    scnn_step<<<gridV, 256, 0, stream>>>(out, wp + 1 * wElems, i, i + 1, W_, 1, W_);
  // Pass 3: left-right (scan W, conv along H)
  for (int j = 1; j < W_; ++j)
    scnn_step<<<gridH, 256, 0, stream>>>(out, wp + 2 * wElems, j, j - 1, H_, W_, 1);
  // Pass 4: right-left
  for (int j = W_ - 2; j >= 0; --j)
    scnn_step<<<gridH, 256, 0, stream>>>(out, wp + 3 * wElems, j, j + 1, H_, W_, 1);
}
